// DenseGATNet_78116865179705
// MI455X (gfx1250) — matrix operand order repaired and checked by reading, not compiled, not run on hardware
//
#include <hip/hip_runtime.h>
#include <hip/hip_bf16.h>

// ---------------------------------------------------------------------------
// DenseGATNet for MI455X (gfx1250, wave32).
// Edge-side GEMMs use v_wmma_f32_16x16x32_bf16; node-side ops are VALU.
// Weights are pre-packed once per launch into exact WMMA fragment layouts
// (bf16, zero-padding baked in) so hot loops do pure b128 fragment loads.
// ---------------------------------------------------------------------------

typedef __attribute__((ext_vector_type(16))) __bf16 v16bf;
typedef __attribute__((ext_vector_type(8)))  float  v8f;
typedef __bf16 bf16_t;

#define NN 256           // nodes per graph
#define BB 2             // batch
#define DX 64
#define DE 32
#define HEADS 3
#define HCH (HEADS*DX)   // 192

__device__ __forceinline__ v8f wmma_bf16(v16bf a, v16bf b, v8f c) {
  return __builtin_amdgcn_wmma_f32_16x16x32_bf16(false, a, false, b, (short)0, c, false, false);
}
// A-matrix (16xK=32 bf16) element->K mapping for lane group kg, element e:
//   K = kg*16 + e   -> one contiguous 16-float run at kg*16 (loaded as two 8-float halves)
// B-matrix (K=32 x N=16) element->K: K = kg*16 + e  -> one contiguous 16-float run

__device__ __forceinline__ void ld8(const float* p, float* d) {
  const float4 x = *(const float4*)p;
  const float4 y = *(const float4*)(p + 4);
  d[0] = x.x; d[1] = x.y; d[2] = x.z; d[3] = x.w;
  d[4] = y.x; d[5] = y.y; d[6] = y.z; d[7] = y.w;
}
__device__ __forceinline__ void a_from_runs(const float* base, int kg, v16bf& a) {
  float r0[8], r1[8];
  ld8(base + kg * 16, r0);
  ld8(base + kg * 16 + 8, r1);
#pragma unroll
  for (int e = 0; e < 8; e++) { a[e] = (__bf16)r0[e]; a[e + 8] = (__bf16)r1[e]; }
}
__device__ __forceinline__ void b_from_contig(const float* base, int kg, v16bf& b) {
  float r0[8], r1[8];
  ld8(base + kg * 16, r0);
  ld8(base + kg * 16 + 8, r1);
#pragma unroll
  for (int e = 0; e < 8; e++) { b[e] = (__bf16)r0[e]; b[e + 8] = (__bf16)r1[e]; }
}
// Load a pre-packed fragment (frag f, this lane): 32 bytes -> 2x global_load_b128
__device__ __forceinline__ v16bf ldfrag(const bf16_t* pack, int f, int lane) {
  return *(const v16bf*)(pack + (((size_t)f * 32 + lane) << 4));
}

// ---------------------------------------------------------------------------
// Weight pre-packing.
// B fragments: frag f = nt*ktiles + kt; element (lane,e) <- W[kt*32+kg*16+e, nt*16+lm]
__global__ void k_pack_B(const float* __restrict__ W, bf16_t* __restrict__ out,
                         int ldn, int Kvalid, int Nvalid, int ktiles, int ntiles) {
  int idx = blockIdx.x * blockDim.x + threadIdx.x;
  int tot = ktiles * ntiles * 512;
  if (idx >= tot) return;
  int e = idx & 15, lane = (idx >> 4) & 31, f = idx >> 9;
  int kt = f % ktiles, nt = f / ktiles;
  int lm = lane & 15, kg = lane >> 4;
  int k = kt * 32 + kg * 16 + e, n = nt * 16 + lm;
  float v = (k < Kvalid && n < Nvalid) ? W[(size_t)k * ldn + n] : 0.f;
  out[idx] = (bf16_t)v;
}
// A-layout fragments of We^T (12 chan tiles): element (lane,e) <- We[akof, ct*16+lm]
__global__ void k_pack_We(const float* __restrict__ We, bf16_t* __restrict__ out) {
  int idx = blockIdx.x * blockDim.x + threadIdx.x;
  if (idx >= 12 * 512) return;
  int e = idx & 15, lane = (idx >> 4) & 31, ct = idx >> 9;
  int lm = lane & 15, kg = lane >> 4;
  int k = (kg << 4) + e;
  out[idx] = (bf16_t)We[k * HCH + ct * 16 + lm];
}
// B fragments of ht with (j,h) K-flattening for the aggregation GEMM.
// frag f = (b*24 + kt)*4 + nt
__global__ void k_pack_ht(const float* __restrict__ ht, bf16_t* __restrict__ out) {
  int idx = blockIdx.x * blockDim.x + threadIdx.x;
  if (idx >= BB * 24 * 4 * 512) return;
  int e = idx & 15, lane = (idx >> 4) & 31, f = idx >> 9;
  int nt = f & 3, kt = (f >> 2) % 24, b = (f >> 2) / 24;
  int lm = lane & 15, kg = lane >> 4;
  int k = kt * 32 + kg * 16 + e;
  int jj = k / HEADS, h = k - jj * HEADS;
  out[idx] = (bf16_t)ht[((size_t)b * NN + jj) * HCH + h * DX + nt * 16 + lm];
}

// ---------------------------------------------------------------------------
// adjem[b,i,j] = (sum_c E[b,i,j,c] != 0) * nm[b,i] * nm[b,j]
__global__ void k_adj_em(const float* __restrict__ E, const float* __restrict__ nm,
                         float* __restrict__ adjem) {
  int idx = blockIdx.x * blockDim.x + threadIdx.x;
  if (idx >= BB * NN * NN) return;
  int j = idx % NN, i = (idx / NN) % NN, b = idx / (NN * NN);
  const float* p = E + (size_t)idx * 8;
  float s = 0.f;
#pragma unroll
  for (int c = 0; c < 8; c++) s += p[c];
  float adj = (s != 0.f) ? 1.f : 0.f;
  adjem[idx] = adj * nm[b * NN + i] * nm[b * NN + j];
}

// generic small node linear: out[M,Nout] = act(in[M,K] @ W[K,Nout] + b)
__global__ void k_node_linear(const float* __restrict__ in, const float* __restrict__ W,
                              const float* __restrict__ bias, float* __restrict__ out,
                              int M, int K, int Nout, int relu) {
  int idx = blockIdx.x * blockDim.x + threadIdx.x;
  if (idx >= M * Nout) return;
  int row = idx / Nout, col = idx % Nout;
  float acc = bias ? bias[col] : 0.f;
  const float* ip = in + (size_t)row * K;
  for (int k = 0; k < K; k++) acc += ip[k] * W[(size_t)k * Nout + col];
  if (relu) acc = fmaxf(acc, 0.f);
  out[idx] = acc;
}

__global__ void k_mask_X(const float* __restrict__ Xh, const float* __restrict__ nm,
                         float* __restrict__ Xm) {
  int idx = blockIdx.x * blockDim.x + threadIdx.x;
  if (idx >= BB * NN * DX) return;
  Xm[idx] = Xh[idx] * nm[idx / DX];
}

__global__ void k_mask_E(const float* __restrict__ Eh, const float* __restrict__ nm,
                         float* __restrict__ Em) {
  int idx = blockIdx.x * blockDim.x + threadIdx.x;
  if (idx >= BB * NN * NN * DE) return;
  int edge = idx / DE;
  int j = edge % NN, i = (edge / NN) % NN, b = edge / (NN * NN);
  Em[idx] = Eh[idx] * nm[b * NN + i] * nm[b * NN + j];
}

// ---------------------------------------------------------------------------
// Fused mlp_in_E: raw E (K=8, zero-padded to 32) -> 128 (relu) -> LDS -> 32 (relu)
__global__ __launch_bounds__(128) void k_inE_fused(
    const float* __restrict__ E, const bf16_t* __restrict__ pkW1, const float* __restrict__ b1,
    const bf16_t* __restrict__ pkW2, const float* __restrict__ b2, float* __restrict__ out) {
  __shared__ float lds[4 * 16 * 128];
  const int lane = threadIdx.x & 31, wv = threadIdx.x >> 5;
  const int lm = lane & 15, kg = lane >> 4;
  const int tile = blockIdx.x * 4 + wv;
  const long edge0 = (long)tile * 16;

  // A: valid K = 0..7 (kg==0, e<8); always load, select value (branchless)
  float av[8];
  ld8(E + (edge0 + lm) * 8, av);
  v16bf a{};
#pragma unroll
  for (int e = 0; e < 8; e++) {
    float v = (kg == 0) ? av[e] : 0.f;
    a[e] = (__bf16)v;
    a[e + 8] = (__bf16)0.f;
  }

  float* myl = lds + wv * 16 * 128;
#pragma unroll
  for (int t = 0; t < 8; t++) {
    int n = t * 16 + lm;
    v16bf bm = ldfrag(pkW1, t, lane);                   // padding baked in
    v8f c{};
    c = wmma_bf16(a, bm, c);
    float bb = b1[n];
#pragma unroll
    for (int r = 0; r < 8; r++) myl[(r + (kg << 3)) * 128 + n] = fmaxf(c[r] + bb, 0.f);
  }
  __syncthreads();

#pragma unroll
  for (int t = 0; t < 2; t++) {
    int n = t * 16 + lm;
    v8f c{};
#pragma unroll
    for (int kt = 0; kt < 4; kt++) {
      v16bf a2{};
      a_from_runs(myl + lm * 128 + kt * 32, kg, a2);    // ds_load_b128 x4
      v16bf bm = ldfrag(pkW2, t * 4 + kt, lane);
      c = wmma_bf16(a2, bm, c);
    }
    float bb = b2[n];
#pragma unroll
    for (int r = 0; r < 8; r++)
      out[(edge0 + r + (kg << 3)) * 32 + n] = fmaxf(c[r] + bb, 0.f);
  }
}

// Eh[b,i,j,c] = 0.5*(T[b,i,j,c] + T[b,j,i,c])
__global__ void k_symmetrize32(const float* __restrict__ T, float* __restrict__ Eh) {
  int idx = blockIdx.x * blockDim.x + threadIdx.x;
  if (idx >= BB * NN * NN * DE) return;
  int c = idx & 31, edge = idx >> 5;
  int j = edge % NN, i = (edge / NN) % NN, b = edge / (NN * NN);
  size_t t = ((((size_t)b * NN + j) * NN + i) << 5) + c;
  Eh[idx] = 0.5f * (T[idx] + T[t]);
}

// ---------------------------------------------------------------------------
// Scores: per wave (b, i, 16-j block). Transposed h_e GEMM (A = We^T pre-packed).
// Fuses leaky_relu(h_s+h_t+h_e) . attn, exp, adj*em mask. h_e never stored.
__global__ __launch_bounds__(256) void k_scores(
    const float* __restrict__ Em, const bf16_t* __restrict__ pkWe,
    const float* __restrict__ hs, const float* __restrict__ ht,
    const float* __restrict__ attn, const float* __restrict__ adjem,
    float* __restrict__ S) {
  const int lane = threadIdx.x & 31, wv = threadIdx.x >> 5;
  const int lm = lane & 15, kg = lane >> 4;
  const int tile = blockIdx.x * 8 + wv;                 // 8192 = bi*16 + jb
  const int jb = tile & 15, bi = tile >> 4;
  const int j = jb * 16 + lm;
  const int b = bi / NN;

  v16bf bm{};
  b_from_contig(Em + (((size_t)bi * NN + j) << 5), kg, bm);   // global_load_b128 x4

  const float* hsp = hs + (size_t)bi * HCH;
  const float* htp = ht + ((size_t)b * NN + j) * HCH;
  float part[HEADS] = {0.f, 0.f, 0.f};

#pragma unroll
  for (int ct = 0; ct < 12; ct++) {
    v16bf a = ldfrag(pkWe, ct, lane);
    v8f c{};
    c = wmma_bf16(a, bm, c);
    const int h = ct >> 2;
#pragma unroll
    for (int r = 0; r < 8; r++) {
      int chan = ct * 16 + r + (kg << 3);
      float v = c[r] + hsp[chan] + htp[chan];
      v = (v > 0.f) ? v : 0.2f * v;                     // leaky_relu(0.2)
      part[h] += v * attn[chan];
    }
  }
  float ae = adjem[(size_t)bi * NN + j];
#pragma unroll
  for (int h = 0; h < HEADS; h++) {
    float full = part[h] + __shfl_xor(part[h], 16, 32);
    float sc = __expf(full) * ae;
    if (kg == 0) S[(((size_t)bi * NN + j) * HEADS) + h] = sc;
  }
}

// per (b,i): rowsum over j per head, then S /= (sum + 1e-6) in place
__global__ void k_rowsum_norm(float* __restrict__ S) {
  __shared__ float red[NN];
  const int bi = blockIdx.x, j = threadIdx.x;
  float* row = S + (size_t)bi * NN * HEADS;
  float v[HEADS];
#pragma unroll
  for (int h = 0; h < HEADS; h++) v[h] = row[j * HEADS + h];
#pragma unroll
  for (int h = 0; h < HEADS; h++) {
    red[j] = v[h];
    __syncthreads();
    for (int s = NN / 2; s > 0; s >>= 1) { if (j < s) red[j] += red[j + s]; __syncthreads(); }
    float sum = red[0];
    __syncthreads();
    row[j * HEADS + h] = v[h] / (sum + 1e-6f);
  }
}

// new_X[b,i,d] = sum_{j,h} Snorm[b,i,j,h] * ht[b,j,h,d]  (ht pre-packed as B frags)
__global__ __launch_bounds__(256) void k_newX(
    const float* __restrict__ S, const bf16_t* __restrict__ pkht, float* __restrict__ nx) {
  const int lane = threadIdx.x & 31, wv = threadIdx.x >> 5;
  const int lm = lane & 15, kg = lane >> 4;
  const int tile = blockIdx.x * 8 + wv;                 // 32 tiles = b*16 + it
  const int b = tile >> 4, it = tile & 15;
  const float* Srow0 = S + (size_t)(b * NN + it * 16) * NN * HEADS;
  v8f c[4] = {};
  for (int kt = 0; kt < 24; kt++) {
    v16bf a{};
    a_from_runs(Srow0 + (size_t)lm * NN * HEADS + kt * 32, kg, a);  // b128 x4
#pragma unroll
    for (int nt = 0; nt < 4; nt++) {
      v16bf bm = ldfrag(pkht, (b * 24 + kt) * 4 + nt, lane);
      c[nt] = wmma_bf16(a, bm, c[nt]);
    }
  }
#pragma unroll
  for (int nt = 0; nt < 4; nt++)
#pragma unroll
    for (int r = 0; r < 8; r++) {
      int i = it * 16 + r + (kg << 3);
      nx[((size_t)b * NN + i) * DX + nt * 16 + lm] = c[nt][r];
    }
}

// edge_mlp: outpre = Em + relu(Em@W3 + Aterm[i] + Bterm[j] + bias)  (pre-LN incl residual)
__global__ __launch_bounds__(256) void k_edgemlp(
    const float* __restrict__ Em, const bf16_t* __restrict__ pkW3, const float* __restrict__ bias,
    const float* __restrict__ At, const float* __restrict__ Bt, float* __restrict__ outpre) {
  const int lane = threadIdx.x & 31, wv = threadIdx.x >> 5;
  const int lm = lane & 15, kg = lane >> 4;
  const int tile = blockIdx.x * 8 + wv;                 // 8192
  const long edge0 = (long)tile * 16;
  const int bi = (int)(edge0 / NN), b = bi / NN, jbase = (int)(edge0 % NN);

  v16bf a{};
  a_from_runs(Em + ((edge0 + lm) << 5), kg, a);         // global_load_b128 x4

#pragma unroll
  for (int t = 0; t < 2; t++) {
    int n = t * 16 + lm;
    v16bf bm = ldfrag(pkW3, t, lane);
    v8f c{};
    c = wmma_bf16(a, bm, c);
    float at = At[(size_t)bi * 32 + n] + bias[n];
#pragma unroll
    for (int r = 0; r < 8; r++) {
      int m = r + (kg << 3), j = jbase + m;
      float v = c[r] + at + Bt[((size_t)b * NN + j) * 32 + n];
      v = fmaxf(v, 0.f);
      outpre[((edge0 + m) << 5) + n] = Em[((edge0 + m) << 5) + n] + v;
    }
  }
}

// LayerNorm over DE channels: out = LN(in)*g + b
__global__ void k_ln_E(const float* __restrict__ in, const float* __restrict__ g,
                       const float* __restrict__ bt, float* __restrict__ out) {
  int e = blockIdx.x * blockDim.x + threadIdx.x;
  if (e >= BB * NN * NN) return;
  const float* p = in + ((size_t)e << 5);
  float mu = 0.f;
  for (int c = 0; c < DE; c++) mu += p[c];
  mu *= (1.f / DE);
  float var = 0.f;
  for (int c = 0; c < DE; c++) { float d = p[c] - mu; var += d * d; }
  var *= (1.f / DE);
  float inv = rsqrtf(var + 1e-5f);
  float* o = out + ((size_t)e << 5);
  for (int c = 0; c < DE; c++) o[c] = (p[c] - mu) * inv * g[c] + bt[c];
}

// X_out = relu( LN(Xm + nx2)*g + b )
__global__ void k_ln_X(const float* __restrict__ Xm, const float* __restrict__ nx2,
                       const float* __restrict__ g, const float* __restrict__ bt,
                       float* __restrict__ out) {
  int r = blockIdx.x * blockDim.x + threadIdx.x;
  if (r >= BB * NN) return;
  float t[DX];
  float mu = 0.f;
  for (int c = 0; c < DX; c++) { t[c] = Xm[(size_t)r * DX + c] + nx2[(size_t)r * DX + c]; mu += t[c]; }
  mu *= (1.f / DX);
  float var = 0.f;
  for (int c = 0; c < DX; c++) { float d = t[c] - mu; var += d * d; }
  var *= (1.f / DX);
  float inv = rsqrtf(var + 1e-5f);
  for (int c = 0; c < DX; c++)
    out[(size_t)r * DX + c] = fmaxf((t[c] - mu) * inv * g[c] + bt[c], 0.f);
}

// ---------------------------------------------------------------------------
// Fused mlp_out_E: Eh (K=32) -> 128 (relu) -> LDS -> 8 (N padded to 16, baked)
__global__ __launch_bounds__(128) void k_outE_fused(
    const float* __restrict__ Ein, const bf16_t* __restrict__ pkW1, const float* __restrict__ b1,
    const bf16_t* __restrict__ pkW2, const float* __restrict__ b2, float* __restrict__ out8) {
  __shared__ float lds[4 * 16 * 128];
  const int lane = threadIdx.x & 31, wv = threadIdx.x >> 5;
  const int lm = lane & 15, kg = lane >> 4;
  const int tile = blockIdx.x * 4 + wv;
  const long edge0 = (long)tile * 16;

  v16bf a{};
  a_from_runs(Ein + ((edge0 + lm) << 5), kg, a);        // global_load_b128 x4

  float* myl = lds + wv * 16 * 128;
#pragma unroll
  for (int t = 0; t < 8; t++) {
    int n = t * 16 + lm;
    v16bf bm = ldfrag(pkW1, t, lane);
    v8f c{};
    c = wmma_bf16(a, bm, c);
    float bb = b1[n];
#pragma unroll
    for (int r = 0; r < 8; r++) myl[(r + (kg << 3)) * 128 + n] = fmaxf(c[r] + bb, 0.f);
  }
  __syncthreads();

  const int n = lm;                                      // only cols 0..7 valid (baked)
  v8f c{};
#pragma unroll
  for (int kt = 0; kt < 4; kt++) {
    v16bf a2{};
    a_from_runs(myl + lm * 128 + kt * 32, kg, a2);      // ds_load_b128 x4
    v16bf bm = ldfrag(pkW2, kt, lane);
    c = wmma_bf16(a2, bm, c);
  }
  if (n < 8) {
    float bb = b2[n];
#pragma unroll
    for (int r = 0; r < 8; r++) out8[(edge0 + r + (kg << 3)) * 8 + n] = c[r] + bb;
  }
}

// Eo = 0.5*(T8[b,i,j,:] + T8[b,j,i,:]) * nm_i * nm_j
__global__ void k_finalE(const float* __restrict__ T8, const float* __restrict__ nm,
                         float* __restrict__ out) {
  int idx = blockIdx.x * blockDim.x + threadIdx.x;
  if (idx >= BB * NN * NN * 8) return;
  int c = idx & 7, edge = idx >> 3;
  int j = edge % NN, i = (edge / NN) % NN, b = edge / (NN * NN);
  size_t t = ((((size_t)b * NN + j) * NN + i) << 3) + c;
  out[idx] = 0.5f * (T8[idx] + T8[t]) * nm[b * NN + i] * nm[b * NN + j];
}

__global__ void k_finalX(const float* __restrict__ Xo, const float* __restrict__ nm,
                         float* __restrict__ out) {
  int idx = blockIdx.x * blockDim.x + threadIdx.x;
  if (idx >= BB * NN * 16) return;
  out[idx] = Xo[idx] * nm[idx / 16];
}

// ---------------------------------------------------------------------------
extern "C" void kernel_launch(void* const* d_in, const int* in_sizes, int n_in,
                              void* d_out, int out_size, void* d_ws, size_t ws_size,
                              hipStream_t stream) {
  (void)in_sizes; (void)n_in; (void)out_size; (void)ws_size;
  // Input order (setup_inputs dict order, params in insertion order, W before b):
  // 0:X 1:E 2:node_mask
  // 3..6   mlp_in_X | 7..10 mlp_in_E | 11..14 mlp_out_X | 15..18 mlp_out_E
  // per layer l, base=19+12*l: Ws,Wt,We,attn,node_W,node_b,edge_W,edge_b,gX,bX,gE,bE
  const float* X  = (const float*)d_in[0];
  const float* E  = (const float*)d_in[1];
  const float* nm = (const float*)d_in[2];
  auto P = [&](int i) { return (const float*)d_in[i]; };

  float* ws    = (float*)d_ws;
  float* Eh    = ws;                       // 4,194,304
  float* Em    = ws + 4194304;             // 4,194,304
  float* Etmp  = ws + 8388608;             // 4,194,304 (also reused as out8 tmp)
  float* Xh    = ws + 12582912;            // 32768
  float* Xm    = Xh + 32768;               // 32768
  float* adjem = Xm + 32768;               // 131072
  float* hs    = adjem + 131072;           // 98304
  float* ht    = hs + 98304;               // 98304
  float* S     = ht + 98304;               // 393216
  float* nx1   = S + 393216;               // 32768
  float* nx2   = nx1 + 32768;              // 32768
  float* At    = nx2 + 32768;              // 16384
  float* Bt    = At + 16384;               // 16384
  float* XoH   = Bt + 16384;               // 131072
  float* Xo    = XoH + 131072;             // 8192
  // bf16 fragment packs (32-byte aligned: all offsets are multiples of 16 elts)
  bf16_t* pk      = (bf16_t*)(Xo + 8192);
  bf16_t* pk_inW1 = pk;                    // 8 frags  = 4096
  bf16_t* pk_inW2 = pk_inW1 + 4096;        // 8 frags  = 4096
  bf16_t* pk_oW1  = pk_inW2 + 4096;        // 8 frags  = 4096
  bf16_t* pk_oW2  = pk_oW1 + 4096;         // 4 frags  = 2048
  bf16_t* pk_We   = pk_oW2 + 2048;         // 12 frags = 6144 (per layer, rewritten)
  bf16_t* pk_W3   = pk_We + 6144;          // 2 frags  = 1024
  bf16_t* pk_ht   = pk_W3 + 1024;          // 192 frags = 98304 (per layer, rewritten)

  auto lin = [&](const float* in, const float* W, const float* b, float* out,
                 int M, int K, int Nn, int relu) {
    int tot = M * Nn;
    k_node_linear<<<(tot + 255) / 256, 256, 0, stream>>>(in, W, b, out, M, K, Nn, relu);
  };
  auto packB = [&](const float* W, bf16_t* out, int ldn, int Kv, int Nv, int kt, int nt) {
    int tot = kt * nt * 512;
    k_pack_B<<<(tot + 255) / 256, 256, 0, stream>>>(W, out, ldn, Kv, Nv, kt, nt);
  };

  const int NEDGE = BB * NN * NN;          // 131072
  const int NROWS = BB * NN;               // 512

  // one-shot packs for the in/out edge MLPs
  packB(P(7),  pk_inW1, 128, 8,   128, 1, 8);   // (8x128)  K padded to 32
  packB(P(9),  pk_inW2, 32,  128, 32,  4, 2);   // (128x32)
  packB(P(15), pk_oW1,  128, 32,  128, 1, 8);   // (32x128)
  packB(P(17), pk_oW2,  8,   128, 8,   4, 1);   // (128x8)  N padded to 16

  // adjacency * edge-mask
  k_adj_em<<<(NEDGE + 255) / 256, 256, 0, stream>>>(E, nm, adjem);

  // input MLPs
  lin(X, P(3), P(4), XoH, NROWS, 16, 256, 1);
  lin(XoH, P(5), P(6), Xh, NROWS, 256, DX, 1);
  k_inE_fused<<<2048, 128, 0, stream>>>(E, pk_inW1, P(8), pk_inW2, P(10), Etmp);
  k_symmetrize32<<<(NEDGE * DE + 255) / 256, 256, 0, stream>>>(Etmp, Eh);

  for (int l = 0; l < 3; l++) {
    int base = 19 + 12 * l;
    const float *Ws = P(base + 0), *Wt = P(base + 1), *We = P(base + 2), *attn = P(base + 3);
    const float *nW = P(base + 4), *nb = P(base + 5), *eW = P(base + 6), *eb = P(base + 7);
    const float *gX = P(base + 8), *bX = P(base + 9), *gE = P(base + 10), *bE = P(base + 11);

    k_pack_We<<<(12 * 512 + 255) / 256, 256, 0, stream>>>(We, pk_We);
    packB(eW + 128 * DE, pk_W3, 32, 32, 32, 1, 2);          // edge_mlp rows 128..159

    k_mask_X<<<(NROWS * DX + 255) / 256, 256, 0, stream>>>(Xh, nm, Xm);
    k_mask_E<<<(NEDGE * DE + 255) / 256, 256, 0, stream>>>(Eh, nm, Em);
    lin(Xm, Ws, nullptr, hs, NROWS, DX, HCH, 0);
    lin(Xm, Wt, nullptr, ht, NROWS, DX, HCH, 0);
    k_pack_ht<<<(BB * 96 * 512 + 255) / 256, 256, 0, stream>>>(ht, pk_ht);
    k_scores<<<1024, 256, 0, stream>>>(Em, pk_We, hs, ht, attn, adjem, S);
    k_rowsum_norm<<<NROWS, NN, 0, stream>>>(S);
    k_newX<<<4, 256, 0, stream>>>(S, pk_ht, nx1);
    lin(nx1, nW, nb, nx2, NROWS, DX, DX, 1);
    lin(nx2, eW, nullptr, At, NROWS, DX, DE, 0);             // rows 0..63  (new_X_i)
    lin(nx2, eW + 64 * DE, nullptr, Bt, NROWS, DX, DE, 0);   // rows 64..127 (new_X_j)
    k_edgemlp<<<1024, 256, 0, stream>>>(Em, pk_W3, eb, At, Bt, Etmp);
    k_ln_E<<<(NEDGE + 255) / 256, 256, 0, stream>>>(Etmp, gE, bE, Eh);
    k_ln_X<<<(NROWS + 255) / 256, 256, 0, stream>>>(Xm, nx2, gX, bX, Xh);  // includes relu
  }

  // output MLPs
  lin(Xh, P(11), P(12), XoH, NROWS, DX, 256, 1);
  lin(XoH, P(13), P(14), Xo, NROWS, 256, 16, 0);
  k_finalX<<<(NROWS * 16 + 255) / 256, 256, 0, stream>>>(Xo, nm, (float*)d_out);

  k_outE_fused<<<2048, 128, 0, stream>>>(Eh, pk_oW1, P(16), pk_oW2, P(18), Etmp);
  k_finalE<<<(NEDGE * 8 + 255) / 256, 256, 0, stream>>>(Etmp, nm, (float*)d_out + NROWS * 16);
}
